// SelfAttention_28063316312367
// MI455X (gfx1250) — matrix operand order repaired and checked by reading, not compiled, not run on hardware
//
#include <hip/hip_runtime.h>

typedef __bf16 bf16;
typedef __attribute__((ext_vector_type(16))) __bf16 v16bf;
typedef __attribute__((ext_vector_type(8)))  __bf16 v8bf;
typedef __attribute__((ext_vector_type(8)))  float  v8f;
typedef int v4i __attribute__((vector_size(16)));

#define DM    1024
#define SEQ   2048
#define BATCH 8
#define RSQ   (1.0f / 32.0f)   // 1/sqrt(1024)

#define BM 128
#define BN 128
#define BK 32
#define LDSS 40                // padded LDS row stride (bf16 elems): 80B, conflict-free

#define AS1 __attribute__((address_space(1)))
#define AS3 __attribute__((address_space(3)))

#if __has_builtin(__builtin_amdgcn_global_load_async_to_lds_b128)
#define USE_ASYNC 1
#endif

// ---------------- staging: global -> LDS (128 rows x 32 cols bf16) ----------------
// 512 chunks of 16B; 256 threads x 2 chunks.
__device__ __forceinline__ void stage_tile(const bf16* __restrict__ g, int ld,
                                           int row0, int k0, bf16* s) {
#pragma unroll
  for (int i = 0; i < 2; ++i) {
    const int c = threadIdx.x + 256 * i;
    const int row = c >> 2, part = (c & 3) * 8;
    const bf16* gp = g + (size_t)(row0 + row) * ld + k0 + part;
    bf16* sp = s + row * LDSS + part;
#ifdef USE_ASYNC
    __builtin_amdgcn_global_load_async_to_lds_b128((AS1 v4i*)gp, (AS3 v4i*)sp, 0, 0);
#else
    *(v8bf*)sp = *(const v8bf*)gp;
#endif
  }
}

__device__ __forceinline__ void wait_stage() {
#ifdef USE_ASYNC
#if __has_builtin(__builtin_amdgcn_s_wait_asynccnt)
  __builtin_amdgcn_s_wait_asynccnt(0);
#else
  asm volatile("s_wait_asynccnt 0x0" ::: "memory");
#endif
#endif
  __syncthreads();
}

// ---------------- WMMA fragment loads from LDS (wave32 16x16x32 bf16) ----------------
// A fragment: element i -> K = 8*h + i (i<8), 16 + 8*h + (i-8) (i>=8); row M fixed per lane.
__device__ __forceinline__ v16bf lds_a_frag(const bf16* s, int row, int h) {
  const bf16* p = s + row * LDSS + 8 * h;
  v16bf a;
#pragma unroll
  for (int i = 0; i < 8; ++i) a[i] = p[i];
#pragma unroll
  for (int i = 0; i < 8; ++i) a[8 + i] = p[16 + i];
  return a;
}
// B fragment from transposed operand tile: column N fixed per lane; K = 8*h + i (i<8), 16 + 8*h + (i-8) (i>=8).
__device__ __forceinline__ v16bf lds_b_frag(const bf16* s, int nrow, int h) {
  const bf16* p = s + nrow * LDSS + 8 * h;
  v16bf b;
#pragma unroll
  for (int i = 0; i < 8; ++i) b[i] = p[i];
#pragma unroll
  for (int i = 0; i < 8; ++i) b[8 + i] = p[16 + i];
  return b;
}
__device__ __forceinline__ v8f wmma_bf16(v16bf a, v16bf b, v8f c) {
  return __builtin_amdgcn_wmma_f32_16x16x32_bf16(false, a, false, b, (short)0, c,
                                                 false, false);
}

// ------------- LDS-tiled GEMM core: C[BMxBN] += A * T^T (T row-major [N][K]) -------------
// 256 threads = 8 waves as 4(M) x 2(N); wave tile 32x64 = 2x4 WMMA tiles.
__device__ __forceinline__ void gemm_core(const bf16* __restrict__ A, int lda,
                                          const bf16* __restrict__ T, int ldt,
                                          int K, int mblk, int nblk,
                                          bf16* sA, bf16* sB, v8f acc[2][4]) {
  const int lane = threadIdx.x & 31;
  const int wave = threadIdx.x >> 5;
  const int l = lane & 15, h = lane >> 4;
  const int mo = (wave >> 1) * 32, no = (wave & 1) * 64;

  for (int k0 = 0; k0 < K; k0 += BK) {
    stage_tile(A, lda, mblk, k0, sA);
    stage_tile(T, ldt, nblk, k0, sB);
    wait_stage();
    v16bf a0 = lds_a_frag(sA, mo + l, h);
    v16bf a1 = lds_a_frag(sA, mo + 16 + l, h);
    v16bf b[4];
#pragma unroll
    for (int j = 0; j < 4; ++j) b[j] = lds_b_frag(sB, no + 16 * j + l, h);
#pragma unroll
    for (int j = 0; j < 4; ++j) {
      acc[0][j] = wmma_bf16(a0, b[j], acc[0][j]);
      acc[1][j] = wmma_bf16(a1, b[j], acc[1][j]);
    }
    __syncthreads();
  }
}

// ---------------- f32 -> bf16 conversion ----------------
__global__ __launch_bounds__(256) void cvt_kernel(const float* __restrict__ src,
                                                  bf16* __restrict__ dst, int n) {
  const int i = (blockIdx.x * 256 + threadIdx.x) * 4;
  if (i < n) {
    const float4 v = *(const float4*)(src + i);
    dst[i + 0] = (__bf16)v.x;
    dst[i + 1] = (__bf16)v.y;
    dst[i + 2] = (__bf16)v.z;
    dst[i + 3] = (__bf16)v.w;
  }
}

// ---------------- Kernel: fused QKV projection (z = 0:Q 1:K 2:V-transposed) ----------------
__global__ __launch_bounds__(256) void qkv_gemm(
    const bf16* __restrict__ Xb, const bf16* __restrict__ Wb,
    const float* __restrict__ bq, const float* __restrict__ bk,
    const float* __restrict__ bv,
    bf16* __restrict__ Qb, bf16* __restrict__ Kb, bf16* __restrict__ Vt) {
  __shared__ bf16 sA[BM * LDSS];
  __shared__ bf16 sB[BN * LDSS];
  const int which = blockIdx.z;
  const bf16* T = Wb + (size_t)which * DM * DM;
  const float* bias = (which == 0) ? bq : (which == 1) ? bk : bv;
  const int mblk = blockIdx.x * BM, nblk = blockIdx.y * BN;

  v8f acc[2][4] = {};
  gemm_core(Xb, DM, T, DM, DM, mblk, nblk, sA, sB, acc);

  const int lane = threadIdx.x & 31, wave = threadIdx.x >> 5;
  const int l = lane & 15, h = lane >> 4;
  const int mo = (wave >> 1) * 32, no = (wave & 1) * 64;
#pragma unroll
  for (int im = 0; im < 2; ++im)
#pragma unroll
    for (int in = 0; in < 4; ++in) {
      const int e = nblk + no + 16 * in + l;
      const float bb = bias[e];
#pragma unroll
      for (int r = 0; r < 8; ++r) {
        const int row = mblk + mo + 16 * im + r + 8 * h;  // global token index
        const float val = acc[im][in][r] + bb;
        if (which == 0) {
          Qb[(size_t)row * DM + e] = (__bf16)val;
        } else if (which == 1) {
          Kb[(size_t)row * DM + e] = (__bf16)val;
        } else {
          const int b = row >> 11, s = row & (SEQ - 1);
          Vt[((size_t)b * DM + e) * SEQ + s] = (__bf16)val;  // V transposed [B][D][S]
        }
      }
    }
}

// ---------------- Kernel: scores = Q K^T / sqrt(D), f32 ----------------
__global__ __launch_bounds__(256) void scores_gemm(const bf16* __restrict__ Qb,
                                                   const bf16* __restrict__ Kb,
                                                   float* __restrict__ Sc) {
  __shared__ bf16 sA[BM * LDSS];
  __shared__ bf16 sB[BN * LDSS];
  const int b = blockIdx.z;
  const int mblk = blockIdx.x * BM, nblk = blockIdx.y * BN;
  const bf16* Q = Qb + (size_t)b * SEQ * DM;
  const bf16* K = Kb + (size_t)b * SEQ * DM;
  float* S = Sc + (size_t)b * SEQ * SEQ;

  v8f acc[2][4] = {};
  gemm_core(Q, DM, K, DM, DM, mblk, nblk, sA, sB, acc);

  const int lane = threadIdx.x & 31, wave = threadIdx.x >> 5;
  const int l = lane & 15, h = lane >> 4;
  const int mo = (wave >> 1) * 32, no = (wave & 1) * 64;
#pragma unroll
  for (int im = 0; im < 2; ++im)
#pragma unroll
    for (int in = 0; in < 4; ++in)
#pragma unroll
      for (int r = 0; r < 8; ++r)
        S[(size_t)(mblk + mo + 16 * im + r + 8 * h) * SEQ + nblk + no + 16 * in + l] =
            acc[im][in][r] * RSQ;
}

// ------- Kernel: per-COLUMN online softmax stats (softmax over the query axis) -------
__global__ __launch_bounds__(256) void colstats_kernel(const float* __restrict__ Sc,
                                                       float* __restrict__ Mcol,
                                                       float* __restrict__ Rz) {
  const int gid = blockIdx.x * 256 + threadIdx.x;  // 0 .. B*SEQ-1
  const int b = gid >> 11, k = gid & (SEQ - 1);
  const float* S = Sc + (size_t)b * SEQ * SEQ + k;
  float m = -INFINITY, z = 0.0f;
  for (int q = 0; q < SEQ; ++q) {
    const float x = S[(size_t)q * SEQ];
    const float mn = fmaxf(m, x);
    z = z * __expf(m - mn) + __expf(x - mn);
    m = mn;
  }
  Mcol[gid] = m;
  Rz[gid] = 1.0f / z;
}

// ---------------- Kernel: attn = exp(s - m) * rZ, bf16 ----------------
__global__ __launch_bounds__(256) void attn_kernel(const float* __restrict__ Sc,
                                                   const float* __restrict__ Mcol,
                                                   const float* __restrict__ Rz,
                                                   bf16* __restrict__ A) {
  const size_t idx = ((size_t)blockIdx.x * 256 + threadIdx.x) * 4;
  const int b = (int)(idx >> 22);                 // / (SEQ*SEQ)
  const int k = (int)(idx & (size_t)(SEQ - 1));   // 4-aligned; same b for all 4
  const int base = b * SEQ + k;
  const float4 s4 = *(const float4*)(Sc + idx);
  const float4 m4 = *(const float4*)(Mcol + base);
  const float4 r4 = *(const float4*)(Rz + base);
  A[idx + 0] = (__bf16)(__expf(s4.x - m4.x) * r4.x);
  A[idx + 1] = (__bf16)(__expf(s4.y - m4.y) * r4.y);
  A[idx + 2] = (__bf16)(__expf(s4.z - m4.z) * r4.z);
  A[idx + 3] = (__bf16)(__expf(s4.w - m4.w) * r4.w);
}

// ---------------- Kernel: out = attn @ V / sqrt(D), f32 ----------------
__global__ __launch_bounds__(256) void av_gemm(const bf16* __restrict__ At,
                                               const bf16* __restrict__ Vt,
                                               float* __restrict__ out) {
  __shared__ bf16 sA[BM * LDSS];
  __shared__ bf16 sB[BN * LDSS];
  const int b = blockIdx.z;
  const int mblk = blockIdx.x * BM, nblk = blockIdx.y * BN;  // m = q, n = d
  const bf16* A = At + (size_t)b * SEQ * SEQ;
  const bf16* V = Vt + (size_t)b * DM * SEQ;                 // [D][S]

  v8f acc[2][4] = {};
  gemm_core(A, SEQ, V, SEQ, SEQ, mblk, nblk, sA, sB, acc);

  const int lane = threadIdx.x & 31, wave = threadIdx.x >> 5;
  const int l = lane & 15, h = lane >> 4;
  const int mo = (wave >> 1) * 32, no = (wave & 1) * 64;
#pragma unroll
  for (int im = 0; im < 2; ++im)
#pragma unroll
    for (int in = 0; in < 4; ++in)
#pragma unroll
      for (int r = 0; r < 8; ++r)
        out[((size_t)b * SEQ + mblk + mo + 16 * im + r + 8 * h) * DM + nblk + no +
            16 * in + l] = acc[im][in][r] * RSQ;
}

// ---------------- Host launcher ----------------
extern "C" void kernel_launch(void* const* d_in, const int* in_sizes, int n_in,
                              void* d_out, int out_size, void* d_ws, size_t ws_size,
                              hipStream_t stream) {
  const float* X  = (const float*)d_in[0];
  const float* Wk = (const float*)d_in[1];
  const float* bk = (const float*)d_in[2];
  const float* Wq = (const float*)d_in[3];
  const float* bq = (const float*)d_in[4];
  const float* Wv = (const float*)d_in[5];
  const float* bv = (const float*)d_in[6];
  float* out = (float*)d_out;

  char* ws = (char*)d_ws;
  size_t off = 0;
  bf16* Xb = (bf16*)(ws + off); off += (size_t)BATCH * SEQ * DM * 2;     // 32 MiB
  bf16* Wb = (bf16*)(ws + off); off += (size_t)3 * DM * DM * 2;          // 6 MiB
  bf16* Qb = (bf16*)(ws + off); off += (size_t)BATCH * SEQ * DM * 2;     // 32 MiB
  bf16* Kb = (bf16*)(ws + off); off += (size_t)BATCH * SEQ * DM * 2;     // 32 MiB
  bf16* Vt = (bf16*)(ws + off); off += (size_t)BATCH * SEQ * DM * 2;     // 32 MiB
  float* Sc = (float*)(ws + off); off += (size_t)BATCH * SEQ * SEQ * 4;  // 128 MiB
  bf16* At = (bf16*)(ws + off); off += (size_t)BATCH * SEQ * SEQ * 2;    // 64 MiB
  float* Mc = (float*)(ws + off); off += (size_t)BATCH * SEQ * 4;
  float* Rz = (float*)(ws + off); off += (size_t)BATCH * SEQ * 4;

  // 0) bf16 conversions (order in Wb matches which: 0=Q, 1=K, 2=V)
  cvt_kernel<<<dim3(BATCH * SEQ * DM / 4 / 256), 256, 0, stream>>>(X, Xb,
                                                                   BATCH * SEQ * DM);
  cvt_kernel<<<dim3(DM * DM / 4 / 256), 256, 0, stream>>>(Wq, Wb + 0 * DM * DM, DM * DM);
  cvt_kernel<<<dim3(DM * DM / 4 / 256), 256, 0, stream>>>(Wk, Wb + 1 * DM * DM, DM * DM);
  cvt_kernel<<<dim3(DM * DM / 4 / 256), 256, 0, stream>>>(Wv, Wb + 2 * DM * DM, DM * DM);
  // 1) QKV projections; V written transposed
  qkv_gemm<<<dim3(BATCH * SEQ / BM, DM / BN, 3), 256, 0, stream>>>(Xb, Wb, bq, bk, bv,
                                                                   Qb, Kb, Vt);
  // 2) scores = Q K^T / 32 (f32; 128 MiB, resident in 192 MB L2)
  scores_gemm<<<dim3(SEQ / BM, SEQ / BN, BATCH), 256, 0, stream>>>(Qb, Kb, Sc);
  // 3) per-column (query-axis) softmax stats
  colstats_kernel<<<dim3(BATCH * SEQ / 256), 256, 0, stream>>>(Sc, Mc, Rz);
  // 4) attention weights in bf16
  attn_kernel<<<dim3((unsigned)((size_t)BATCH * SEQ * SEQ / 4 / 256)), 256, 0, stream>>>(
      Sc, Mc, Rz, At);
  // 5) out = attn V / 32
  av_gemm<<<dim3(SEQ / BM, DM / BN, BATCH), 256, 0, stream>>>(At, Vt, out);
}
